// SparseAttention_53687091200282
// MI455X (gfx1250) — matrix operand order repaired and checked by reading, not compiled, not run on hardware
//
#include <hip/hip_runtime.h>

typedef __attribute__((ext_vector_type(16))) __bf16        v16bf;
typedef __attribute__((ext_vector_type(8)))  float         v8f;
typedef __attribute__((ext_vector_type(8)))  unsigned int  v8u;
typedef __attribute__((ext_vector_type(4)))  int           v4i;

__device__ __forceinline__ unsigned short f2bf(float f) {
    unsigned int u = __builtin_bit_cast(unsigned int, f);
    return (unsigned short)((u + 0x7FFFu + ((u >> 16) & 1u)) >> 16);
}
__device__ __forceinline__ unsigned int pack_bf2(float lo, float hi) {
    return (unsigned int)f2bf(lo) | ((unsigned int)f2bf(hi) << 16);
}
__device__ __forceinline__ v8f wmma_bf16(v16bf a, v16bf b, v8f c) {
    return __builtin_amdgcn_wmma_f32_16x16x32_bf16(false, a, false, b, (short)0, c, false, false);
}

// --------------------- async global->LDS (CDNA5 ASYNCcnt path) --------------
#if __has_builtin(__builtin_amdgcn_global_load_async_to_lds_b128)
#define HAVE_ASYNC 1
#else
#define HAVE_ASYNC 0
#endif

typedef __attribute__((address_space(1))) v4i gas_v4i;   // global int4
typedef __attribute__((address_space(3))) v4i las_v4i;   // LDS int4

__device__ __forceinline__ void cp16(const unsigned short* g, unsigned short* l) {
#if HAVE_ASYNC
    __builtin_amdgcn_global_load_async_to_lds_b128(
        (gas_v4i*)g, (las_v4i*)l, 0, 0);
#else
    *(uint4*)l = *(const uint4*)g;
#endif
}
#if HAVE_ASYNC
#define ASYNC_WAIT(n) asm volatile("s_wait_asynccnt %0" :: "i"(n) : "memory")
#else
#define ASYNC_WAIT(n)
#endif

// DPP 16-lane butterfly reductions (rows of 16 = exactly our C-layout half-wave)
#define DPP_STEP(x, ctrl, OPER)                                                       \
    x = OPER(x, __builtin_bit_cast(float, __builtin_amdgcn_mov_dpp(                   \
               __builtin_bit_cast(int, x), (ctrl), 0xF, 0xF, true)))
__device__ __forceinline__ float red_max16(float x) {
    DPP_STEP(x, 0xB1, fmaxf);   // quad_perm [1,0,3,2]
    DPP_STEP(x, 0x4E, fmaxf);   // quad_perm [2,3,0,1]
    DPP_STEP(x, 0x141, fmaxf);  // row_half_mirror
    DPP_STEP(x, 0x140, fmaxf);  // row_mirror
    return x;
}
__device__ __forceinline__ float adds(float a, float b) { return a + b; }
__device__ __forceinline__ float red_sum16(float x) {
    DPP_STEP(x, 0xB1, adds);
    DPP_STEP(x, 0x4E, adds);
    DPP_STEP(x, 0x141, adds);
    DPP_STEP(x, 0x140, adds);
    return x;
}

#define ST 72   // LDS row stride in bf16: 144B (16B aligned), 36 banks -> conflict-free

// A-layout K index for vgpr-pair v (0..7) within a 32-wide contraction chunk
#define AKIDX(v, hi) ((hi) * 16 + (v) * 2)

// ---------------------------------------------------------------------------
// fp32 -> bf16 bulk conversion (one-shot, bandwidth-trivial)
// ---------------------------------------------------------------------------
__global__ __launch_bounds__(256) void cvt_bf16(const float* __restrict__ src,
                                                unsigned short* __restrict__ dst,
                                                int n8) {
    int i = blockIdx.x * 256 + threadIdx.x;
    if (i < n8) {
        float4 a = ((const float4*)src)[2 * i];
        float4 b = ((const float4*)src)[2 * i + 1];
        uint4 o;
        o.x = pack_bf2(a.x, a.y); o.y = pack_bf2(a.z, a.w);
        o.z = pack_bf2(b.x, b.y); o.w = pack_bf2(b.z, b.w);
        ((uint4*)dst)[i] = o;
    }
}

// ---------------------------------------------------------------------------
// GEMM: C[M,N] = A[M,1024] @ W[N,1024]^T, A and W pre-converted bf16.
// Tiles 128x64, BK=64, double-buffered LDS, async staging (12 ops/thread).
// 128 threads = 4 waves; each wave owns a 32x64 strip (2x4 accum tiles).
// EPI 0: fused RoPE/mask epilogue -> Q,K (rows) and V (transposed) bf16.
// EPI 1: mask -> fp32 out.
// ---------------------------------------------------------------------------
template <int EPI>
__global__ __launch_bounds__(128) void gemm128x64(
    const unsigned short* __restrict__ A, const unsigned short* __restrict__ W,
    const float* __restrict__ cosT, const float* __restrict__ sinT,
    const int* __restrict__ amask,
    unsigned short* __restrict__ Qb, unsigned short* __restrict__ Kb,
    unsigned short* __restrict__ Vb, float* __restrict__ Cout)
{
    __shared__ __align__(16) unsigned short As[2][128 * ST];
    __shared__ __align__(16) unsigned short Bs[2][64 * ST];
    const int tid  = threadIdx.x;
    const int lane = tid & 31, w = tid >> 5;           // 4 waves
    const int nb = blockIdx.x, mb = blockIdx.y;
    const int row0 = mb * 128, col0 = nb * 64;
    const int m = lane & 15, hi = lane >> 4;

    auto stage = [&](int k0, int bb) {
#pragma unroll
        for (int i = 0; i < 8; ++i) {                  // A tile 128x64 bf16
            int idx = tid + i * 128;
            int r = idx >> 3, c = (idx & 7) * 8;
            cp16(A + (size_t)(row0 + r) * 1024 + k0 + c, &As[bb][r * ST + c]);
        }
#pragma unroll
        for (int i = 0; i < 4; ++i) {                  // W tile 64x64 bf16
            int idx = tid + i * 128;
            int r = idx >> 3, c = (idx & 7) * 8;
            cp16(W + (size_t)(col0 + r) * 1024 + k0 + c, &Bs[bb][r * ST + c]);
        }
    };

    v8f acc[2][4] = {};

    stage(0, 0);
    for (int it = 0; it < 16; ++it) {
        const int bb = it & 1;
        if (it + 1 < 16) { stage((it + 1) * 64, bb ^ 1); ASYNC_WAIT(12); }
        else             { ASYNC_WAIT(0); }
        __syncthreads();                               // buffer bb ready

#pragma unroll
        for (int dc = 0; dc < 2; ++dc) {               // 32-wide contraction chunks
            v16bf af[2];
#pragma unroll
            for (int r = 0; r < 2; ++r) {
                v8u au;
                const unsigned short* ab = &As[bb][(w * 32 + r * 16 + m) * ST + dc * 32];
#pragma unroll
                for (int v = 0; v < 8; ++v)
                    au[v] = *(const unsigned int*)(ab + (AKIDX(v, hi)));
                af[r] = __builtin_bit_cast(v16bf, au);
            }
#pragma unroll
            for (int t = 0; t < 4; ++t) {
                v8u bu;
                const unsigned short* bp = &Bs[bb][(t * 16 + m) * ST + dc * 32 + hi * 16];
#pragma unroll
                for (int v = 0; v < 8; ++v)
                    bu[v] = *(const unsigned int*)(bp + v * 2);
                v16bf bf = __builtin_bit_cast(v16bf, bu);
                acc[0][t] = wmma_bf16(af[0], bf, acc[0][t]);
                acc[1][t] = wmma_bf16(af[1], bf, acc[1][t]);
            }
        }
        __syncthreads();                               // reads of buffer bb done
    }

    // 0.125 * log2(e): softmax runs in exp2 domain
    const float QSC = 0.1803368801111244f;
#pragma unroll
    for (int r = 0; r < 2; ++r) {
        if (EPI == 0) {
            const int h48 = nb;                        // one head per N-block
#pragma unroll
            for (int j = 0; j < 8; ++j) {
                int grow = row0 + w * 32 + r * 16 + hi * 8 + j;
                int b = grow >> 11, l = grow & 2047;
                if (h48 < 32) {                        // q or k: RoPE
#pragma unroll
                    for (int t = 0; t < 2; ++t) {
                        int d1 = t * 16 + m, d2 = d1 + 32;
                        float x1 = acc[r][t][j], x2 = acc[r][t + 2][j];
                        float r1 = x1 * cosT[l * 64 + d1] - x2 * sinT[l * 64 + d1];
                        float r2 = x2 * cosT[l * 64 + d2] + x1 * sinT[l * 64 + d2];
                        if (h48 < 16) {
                            float sc = QSC * (amask[b * 2048 + l] ? 1.0f : 0.0f);
                            size_t base = ((size_t)(b * 16 + h48) * 2048 + l) * 64;
                            Qb[base + d1] = f2bf(r1 * sc);
                            Qb[base + d2] = f2bf(r2 * sc);
                        } else {
                            size_t base = ((size_t)(b * 16 + (h48 - 16)) * 2048 + l) * 64;
                            Kb[base + d1] = f2bf(r1);
                            Kb[base + d2] = f2bf(r2);
                        }
                    }
                } else {                               // v: store TRANSPOSED [bh][d][l]
                    size_t base = (size_t)(b * 16 + (h48 - 32)) * 64 * 2048 + l;
#pragma unroll
                    for (int t = 0; t < 4; ++t)
                        Vb[base + (size_t)(t * 16 + m) * 2048] = f2bf(acc[r][t][j]);
                }
            }
        } else {
#pragma unroll
            for (int j = 0; j < 8; ++j) {
                int grow = row0 + w * 32 + r * 16 + hi * 8 + j;
                int b = grow >> 11, l = grow & 2047;
                float msk = amask[b * 2048 + l] ? 1.0f : 0.0f;
#pragma unroll
                for (int t = 0; t < 4; ++t)
                    Cout[(size_t)grow * 1024 + col0 + t * 16 + m] = acc[r][t][j] * msk;
            }
        }
    }
}

// ---------------------------------------------------------------------------
// Flash attention: block = (b,h, 64-query tile), 4 waves x 16 rows, KV chunk 64.
// K/V double-buffered in LDS via async global->LDS copies (ASYNCcnt).
// Softmax in exp2 domain (log2e pre-folded into Q).
// ---------------------------------------------------------------------------
__global__ __launch_bounds__(128) void attn64(
    const unsigned short* __restrict__ Qb, const unsigned short* __restrict__ Kb,
    const unsigned short* __restrict__ Vb, unsigned short* __restrict__ Ob)
{
    __shared__ __align__(16) unsigned short Ks[2][64 * ST];
    __shared__ __align__(16) unsigned short Vt[2][64 * ST];
    __shared__ __align__(16) unsigned short Pw[4 * 16 * ST];

    const int tid  = threadIdx.x;
    const int lane = tid & 31, w = tid >> 5;
    const int m = lane & 15, hi = lane >> 4;
    const int bh = blockIdx.y, qt = blockIdx.x;
    const size_t hbase = (size_t)bh * 2048 * 64;       // K: [bh][l][d]
    const size_t vbase = (size_t)bh * 64 * 2048;       // V: [bh][d][l]
    const int q0 = qt * 64 + w * 16;

    auto stage = [&](int kc, int bb) {
#pragma unroll
        for (int i = 0; i < 4; ++i) {
            int idx = tid + i * 128;                    // 512 x 16B per matrix
            int r = idx >> 3, c = (idx & 7) * 8;
            cp16(Kb + hbase + (size_t)(kc + r) * 64 + c, &Ks[bb][r * ST + c]);
            cp16(Vb + vbase + (size_t)r * 2048 + kc + c, &Vt[bb][r * ST + c]);
        }
    };

    v16bf qf[2];                                        // Q A-fragments (d 0..31, 32..63)
#pragma unroll
    for (int c = 0; c < 2; ++c) {
        v8u u;
        const unsigned short* qp = Qb + hbase + (size_t)(q0 + m) * 64 + c * 32;
#pragma unroll
        for (int v = 0; v < 8; ++v)
            u[v] = *(const unsigned int*)(qp + (AKIDX(v, hi)));
        qf[c] = __builtin_bit_cast(v16bf, u);
    }

    float mrow[8], lrow[8];
    v8f o[4] = {};
#pragma unroll
    for (int j = 0; j < 8; ++j) { mrow[j] = -3.0e38f; lrow[j] = 0.f; }

    unsigned short* Pmine = &Pw[w * 16 * ST];

    stage(0, 0);
    for (int it = 0; it < 32; ++it) {
        const int bb = it & 1;
        if (it + 1 < 32) { stage((it + 1) * 64, bb ^ 1); ASYNC_WAIT(8); }
        else             { ASYNC_WAIT(0); }
        __syncthreads();                                // buffer bb ready for all waves

        v8f s[4] = {};                                  // S = Q @ K^T (16x64)
#pragma unroll
        for (int nt = 0; nt < 4; ++nt)
#pragma unroll
            for (int dc = 0; dc < 2; ++dc) {
                v8u bu;
                const unsigned short* kp = &Ks[bb][(nt * 16 + m) * ST + dc * 32 + hi * 16];
#pragma unroll
                for (int v = 0; v < 8; ++v)
                    bu[v] = *(const unsigned int*)(kp + v * 2);
                s[nt] = wmma_bf16(qf[dc], __builtin_bit_cast(v16bf, bu), s[nt]);
            }

        float ef[8];                                    // online softmax per row
#pragma unroll
        for (int j = 0; j < 8; ++j) {
            float mx = fmaxf(fmaxf(s[0][j], s[1][j]), fmaxf(s[2][j], s[3][j]));
            mx = red_max16(mx);
            float nm = fmaxf(mrow[j], mx);
            ef[j] = exp2f(mrow[j] - nm);
            mrow[j] = nm;
            float rs = 0.f;
#pragma unroll
            for (int nt = 0; nt < 4; ++nt) {
                float p = exp2f(s[nt][j] - nm);
                s[nt][j] = p;
                rs += p;
            }
            rs = red_sum16(rs);
            lrow[j] = lrow[j] * ef[j] + rs;
        }
#pragma unroll
        for (int t = 0; t < 4; ++t)
#pragma unroll
            for (int j = 0; j < 8; ++j)
                o[t][j] *= ef[j];

#pragma unroll
        for (int nt = 0; nt < 4; ++nt)                  // P: C-layout -> LDS
#pragma unroll
            for (int j = 0; j < 8; ++j)
                Pmine[(hi * 8 + j) * ST + nt * 16 + m] = f2bf(s[nt][j]);
        __syncthreads();

#pragma unroll
        for (int c = 0; c < 2; ++c) {                   // O += P @ V (keys c*32..)
            v8u pu;
#pragma unroll
            for (int v = 0; v < 8; ++v)
                pu[v] = *(const unsigned int*)(Pmine + m * ST + c * 32 + (AKIDX(v, hi)));
            v16bf pf = __builtin_bit_cast(v16bf, pu);
#pragma unroll
            for (int t = 0; t < 4; ++t) {
                v8u vu;
                const unsigned short* vp = &Vt[bb][(t * 16 + m) * ST + c * 32 + hi * 16];
#pragma unroll
                for (int v = 0; v < 8; ++v)
                    vu[v] = *(const unsigned int*)(vp + v * 2);
                o[t] = wmma_bf16(pf, __builtin_bit_cast(v16bf, vu), o[t]);
            }
        }
        __syncthreads();                                // all reads of buffer bb done
    }

    const int b = bh >> 4, h = bh & 15;
#pragma unroll
    for (int j = 0; j < 8; ++j) {
        float inv = 1.0f / lrow[j];
        int qr = q0 + hi * 8 + j;
        size_t orow = ((size_t)(b * 2048 + qr)) * 1024 + h * 64;
#pragma unroll
        for (int t = 0; t < 4; ++t)
            Ob[orow + t * 16 + m] = f2bf(o[t][j] * inv);
    }
}

// ---------------------------------------------------------------------------
extern "C" void kernel_launch(void* const* d_in, const int* in_sizes, int n_in,
                              void* d_out, int out_size, void* d_ws, size_t ws_size,
                              hipStream_t stream) {
    const float* hidden = (const float*)d_in[0];
    const float* cosT   = (const float*)d_in[1];
    const float* sinT   = (const float*)d_in[2];
    const int*   amask  = (const int*)d_in[3];
    const float* Wqkv   = (const float*)d_in[4];
    const float* Wo     = (const float*)d_in[5];
    float*       out    = (float*)d_out;

    const size_t seg = (size_t)8192 * 1024;             // elements per bf16 buffer
    unsigned short* Qb   = (unsigned short*)d_ws;
    unsigned short* Kb   = Qb + seg;
    unsigned short* Vb   = Kb + seg;
    unsigned short* Ob   = Vb + seg;
    unsigned short* Hbf  = Ob + seg;                    // hidden bf16  (8192x1024)
    unsigned short* Wqb  = Hbf + seg;                   // Wqkv bf16    (3072x1024)
    unsigned short* Wob  = Wqb + (size_t)3072 * 1024;   // Wo bf16      (1024x1024)

    cvt_bf16<<<dim3(4096), dim3(256), 0, stream>>>(hidden, Hbf, 8192 * 1024 / 8);
    cvt_bf16<<<dim3(1536), dim3(256), 0, stream>>>(Wqkv, Wqb, 3072 * 1024 / 8);
    cvt_bf16<<<dim3(512),  dim3(256), 0, stream>>>(Wo,   Wob, 1024 * 1024 / 8);

    gemm128x64<0><<<dim3(48, 64), dim3(128), 0, stream>>>(
        Hbf, Wqb, cosT, sinT, amask, Qb, Kb, Vb, nullptr);
    attn64<<<dim3(32, 64), dim3(128), 0, stream>>>(Qb, Kb, Vb, Ob);
    gemm128x64<1><<<dim3(16, 64), dim3(128), 0, stream>>>(
        Ob, Wob, cosT, sinT, amask, nullptr, nullptr, nullptr, out);
}
